// DecoderRNN_47768626266744
// MI455X (gfx1250) — compile-verified, matrix operand order checked by reading, not run on hardware
//
#include <hip/hip_runtime.h>
#include <hip/hip_bf16.h>

typedef _Float16 v16h __attribute__((ext_vector_type(16)));
typedef _Float16 v8h  __attribute__((ext_vector_type(8)));
typedef float    v8f  __attribute__((ext_vector_type(8)));

#define B_ 128
#define S_ 20
#define T_ 19
#define E_ 512
#define H_ 512
#define V_ 10000
#define G_ 2048   // 4*H

// ---------------------------------------------------------------------------
// Operand fragment load per CDNA5 16-bit A/B layout:
// lanes 0-15 hold K {0..7,16..23} of their row; lanes 16-31 hold K {8..15,24..31}.
// Two contiguous 16-byte chunks -> global_load_b128 pair.
// ---------------------------------------------------------------------------
__device__ __forceinline__ v16h ldfrag(const _Float16* __restrict__ p, int hi) {
    v16h v;
    v8h* vp = reinterpret_cast<v8h*>(&v);
    vp[0] = *reinterpret_cast<const v8h*>(p + hi * 8);
    vp[1] = *reinterpret_cast<const v8h*>(p + 16 + hi * 8);
    return v;
}

__device__ __forceinline__ v8f wmma32(v16h a, v16h b, v8f c) {
    return __builtin_amdgcn_wmma_f32_16x16x32_f16(false, a, false, b,
                                                  (short)0, c, false, false);
}

// 2x2 tile-block accumulator: 32x32 output region per wave.
// A fragments shared across 2 B fragments and vice versa -> 2 loads/WMMA.
struct Acc22 { v8f a00, a01, a10, a11; };

// Bsrc is (N x K) row-major (B-matrix column n = Bsrc row n).
// n1off: row offset of the second n-tile (16 normally; 0 to alias when the
// second tile would be out of range -- computed but not stored).
__device__ __forceinline__ void gemm22(const _Float16* __restrict__ A,
                                       const _Float16* __restrict__ Bsrc,
                                       int m0, int n0, int lda, int ldb,
                                       int K, int n1off, Acc22& acc) {
    const int lane = threadIdx.x & 31;
    const int hi = lane >> 4, lr = lane & 15;
    const _Float16* a0 = A + (size_t)(m0 + lr) * lda;
    const _Float16* a1 = a0 + (size_t)16 * lda;
    const _Float16* b0 = Bsrc + (size_t)(n0 + lr) * ldb;
    const _Float16* b1 = b0 + (size_t)n1off * ldb;
#pragma unroll
    for (int k0 = 0; k0 < K; k0 += 32) {
        v16h A0 = ldfrag(a0 + k0, hi);
        v16h A1 = ldfrag(a1 + k0, hi);
        v16h B0 = ldfrag(b0 + k0, hi);
        v16h B1 = ldfrag(b1 + k0, hi);
        acc.a00 = wmma32(A0, B0, acc.a00);
        acc.a10 = wmma32(A1, B0, acc.a10);
        acc.a01 = wmma32(A0, B1, acc.a01);
        acc.a11 = wmma32(A1, B1, acc.a11);
    }
}

// ---------------------------------------------------------------------------
// f32 -> f16 weight conversion
// ---------------------------------------------------------------------------
__global__ void f32_to_f16_kernel(const float* __restrict__ src,
                                  _Float16* __restrict__ dst, int n) {
    int i = blockIdx.x * blockDim.x + threadIdx.x;
    if (i < n) dst[i] = (_Float16)src[i];
}

// ---------------------------------------------------------------------------
// Prep: stable descending argsort of 128 lengths, side outputs, zero h/c.
// ---------------------------------------------------------------------------
__global__ void prep_kernel(const int* __restrict__ length,
                            const int* __restrict__ captions,
                            int* __restrict__ sort_ind,
                            int* __restrict__ dlen_i,
                            float* __restrict__ out_tail,  // caps(B*S), dlen(B), sidx(B)
                            _Float16* __restrict__ h16,
                            float* __restrict__ c) {
    int b = threadIdx.x;                 // one block of 128 threads
    int myl = length[b];
    int rank = 0;
    for (int j = 0; j < B_; ++j) {
        int lj = length[j];
        if (lj > myl || (lj == myl && j < b)) rank++;
    }
    sort_ind[rank] = b;
    __syncthreads();
    int si = sort_ind[b];                // sorted position b
    int dl = length[si] - 1;
    dlen_i[b] = dl;
    out_tail[B_ * S_ + b]      = (float)dl;
    out_tail[B_ * S_ + B_ + b] = (float)si;
    for (int s = 0; s < S_; ++s)
        out_tail[b * S_ + s] = (float)captions[si * S_ + s];
    for (int k = b; k < B_ * H_; k += B_) {
        h16[k] = (_Float16)0.f;
        c[k]   = 0.f;
    }
}

// ---------------------------------------------------------------------------
// Gather step inputs: x[0]=images[sort], x[t>0]=emb[captions_s[:,t-1]]
// ---------------------------------------------------------------------------
__global__ void gather_x_kernel(const float* __restrict__ images,
                                const int* __restrict__ captions,
                                const float* __restrict__ emb,
                                const int* __restrict__ sort_ind,
                                _Float16* __restrict__ x16) {
    int idx = blockIdx.x * blockDim.x + threadIdx.x;   // 20*128*512 threads
    int t   = idx / (B_ * E_);
    int rem = idx % (B_ * E_);
    int p   = rem / E_;
    int e   = rem % E_;
    int si  = sort_ind[p];
    float v;
    if (t == 0) v = images[(size_t)si * E_ + e];
    else        v = emb[(size_t)captions[si * S_ + (t - 1)] * E_ + e];
    x16[idx] = (_Float16)v;
}

// ---------------------------------------------------------------------------
// Gates GEMM: gates(128x2048) = x*W_ih^T + h*W_hh^T + (b_ih+b_hh)
// 2x2 tiles per wave: 4 m-pairs x 64 n-pairs = 256 wave-tiles -> 32 blocks.
// ---------------------------------------------------------------------------
__global__ void lstm_gates_kernel(const _Float16* __restrict__ x,
                                  const _Float16* __restrict__ h,
                                  const _Float16* __restrict__ Wih,
                                  const _Float16* __restrict__ Whh,
                                  const float* __restrict__ b_ih,
                                  const float* __restrict__ b_hh,
                                  float* __restrict__ gates) {
    int wave = threadIdx.x >> 5;
    int tile = blockIdx.x * 8 + wave;     // 0..255
    int m0 = (tile & 3) * 32;
    int n0 = (tile >> 2) * 32;
    Acc22 acc = {};
    gemm22(x, Wih, m0, n0, E_, E_, E_, 16, acc);
    gemm22(h, Whh, m0, n0, H_, H_, H_, 16, acc);
    int lane = threadIdx.x & 31;
    int hi = lane >> 4, lr = lane & 15;
    int nA = n0 + lr, nB = n0 + 16 + lr;
    float bbA = b_ih[nA] + b_hh[nA];
    float bbB = b_ih[nB] + b_hh[nB];
#pragma unroll
    for (int r = 0; r < 8; ++r) {
        int mA = m0 + r + hi * 8;
        int mB = mA + 16;
        gates[(size_t)mA * G_ + nA] = acc.a00[r] + bbA;
        gates[(size_t)mA * G_ + nB] = acc.a01[r] + bbB;
        gates[(size_t)mB * G_ + nA] = acc.a10[r] + bbA;
        gates[(size_t)mB * G_ + nB] = acc.a11[r] + bbB;
    }
}

// ---------------------------------------------------------------------------
// LSTM cell elementwise: c' = sig(f)*c + sig(i)*tanh(g); h = sig(o)*tanh(c')
// ---------------------------------------------------------------------------
__device__ __forceinline__ float sigm(float x) { return 1.f / (1.f + __expf(-x)); }

__global__ void lstm_cell_kernel(const float* __restrict__ gates,
                                 float* __restrict__ c,
                                 _Float16* __restrict__ h16) {
    int idx = blockIdx.x * blockDim.x + threadIdx.x;   // B*H threads
    int b = idx >> 9;          // /512
    int n = idx & (H_ - 1);
    const float* g = gates + (size_t)b * G_;
    float gi = g[n], gf = g[H_ + n], gg = g[2 * H_ + n], go = g[3 * H_ + n];
    float cn = sigm(gf) * c[idx] + sigm(gi) * tanhf(gg);
    c[idx] = cn;
    h16[idx] = (_Float16)(sigm(go) * tanhf(cn));
}

// ---------------------------------------------------------------------------
// Classifier GEMM: preds(128x10000) = h*fc_w^T + fc_b, masked by dec_len>t,
// written straight to d_out at [m, t, :].
// 2x2 tiles per wave: 4 m-pairs x 313 n-pairs = 1252 wave-tiles -> 157 blocks.
// Last n-pair aliases its second B tile (computed, not stored).
// ---------------------------------------------------------------------------
__global__ void fc_kernel(const _Float16* __restrict__ h16,
                          const _Float16* __restrict__ fcw,
                          const float* __restrict__ fcb,
                          const int* __restrict__ dec_len,
                          int t, float* __restrict__ out) {
    int wave = threadIdx.x >> 5;
    int tile = blockIdx.x * 8 + wave;
    if (tile >= 4 * 313) return;          // wave-uniform exit, before any WMMA
    int m0 = (tile & 3) * 32;
    int n0 = (tile >> 2) * 32;
    bool n1v = (n0 + 16) < V_;            // second n-tile valid?
    Acc22 acc = {};
    gemm22(h16, fcw, m0, n0, H_, H_, H_, n1v ? 16 : 0, acc);
    int lane = threadIdx.x & 31;
    int hi = lane >> 4, lr = lane & 15;
    int nA = n0 + lr, nB = n0 + 16 + lr;
    float biasA = fcb[nA];
    float biasB = n1v ? fcb[nB] : 0.f;
#pragma unroll
    for (int r = 0; r < 8; ++r) {
        int mA = m0 + r + hi * 8;
        int mB = mA + 16;
        bool kA = dec_len[mA] > t;
        bool kB = dec_len[mB] > t;
        out[((size_t)mA * T_ + t) * V_ + nA] = kA ? acc.a00[r] + biasA : 0.f;
        out[((size_t)mB * T_ + t) * V_ + nA] = kB ? acc.a10[r] + biasA : 0.f;
        if (n1v) {
            out[((size_t)mA * T_ + t) * V_ + nB] = kA ? acc.a01[r] + biasB : 0.f;
            out[((size_t)mB * T_ + t) * V_ + nB] = kB ? acc.a11[r] + biasB : 0.f;
        }
    }
}

// ---------------------------------------------------------------------------
extern "C" void kernel_launch(void* const* d_in, const int* in_sizes, int n_in,
                              void* d_out, int out_size, void* d_ws, size_t ws_size,
                              hipStream_t stream) {
    const float* images   = (const float*)d_in[0];
    const int*   captions = (const int*)  d_in[1];
    const int*   length   = (const int*)  d_in[2];
    const float* emb      = (const float*)d_in[3];
    const float* W_ih     = (const float*)d_in[4];
    const float* W_hh     = (const float*)d_in[5];
    const float* b_ih     = (const float*)d_in[6];
    const float* b_hh     = (const float*)d_in[7];
    const float* fc_w     = (const float*)d_in[8];
    const float* fc_b     = (const float*)d_in[9];
    float* out = (float*)d_out;

    char* ws = (char*)d_ws;
    size_t off = 0;
    auto carve = [&](size_t bytes) -> char* {
        char* p = ws + off;
        off = (off + bytes + 255) & ~(size_t)255;
        return p;
    };
    int*      sort_ind = (int*)     carve(B_ * sizeof(int));
    int*      dlen_i   = (int*)     carve(B_ * sizeof(int));
    _Float16* Wih16    = (_Float16*)carve((size_t)G_ * E_ * 2);
    _Float16* Whh16    = (_Float16*)carve((size_t)G_ * H_ * 2);
    _Float16* fcw16    = (_Float16*)carve((size_t)V_ * H_ * 2);
    _Float16* x16      = (_Float16*)carve((size_t)S_ * B_ * E_ * 2);
    _Float16* h16      = (_Float16*)carve((size_t)B_ * H_ * 2);
    float*    c        = (float*)   carve((size_t)B_ * H_ * 4);
    float*    gates    = (float*)   carve((size_t)B_ * G_ * 4);
    (void)ws_size; (void)n_in; (void)in_sizes; (void)out_size;

    float* out_tail = out + (size_t)B_ * T_ * V_;  // caps, dec_len, sort_ind

    // 1) weight conversion to f16
    f32_to_f16_kernel<<<(G_ * E_) / 256, 256, 0, stream>>>(W_ih, Wih16, G_ * E_);
    f32_to_f16_kernel<<<(G_ * H_) / 256, 256, 0, stream>>>(W_hh, Whh16, G_ * H_);
    f32_to_f16_kernel<<<(V_ * H_) / 256, 256, 0, stream>>>(fc_w, fcw16, V_ * H_);

    // 2) sort + side outputs + zero state
    prep_kernel<<<1, B_, 0, stream>>>(length, captions, sort_ind, dlen_i,
                                      out_tail, h16, c);

    // 3) gather all step inputs
    gather_x_kernel<<<(S_ * B_ * E_) / 256, 256, 0, stream>>>(
        images, captions, emb, sort_ind, x16);

    // 4) sequential LSTM: step 0 consumes features; steps 1..19 also emit preds
    for (int t = 0; t < S_; ++t) {
        const _Float16* x_t = x16 + (size_t)t * B_ * E_;
        lstm_gates_kernel<<<32, 256, 0, stream>>>(x_t, h16, Wih16, Whh16,
                                                  b_ih, b_hh, gates);
        lstm_cell_kernel<<<(B_ * H_) / 256, 256, 0, stream>>>(gates, c, h16);
        if (t >= 1) {
            fc_kernel<<<157, 256, 0, stream>>>(h16, fcw16, fc_b, dlen_i,
                                               t - 1, out);
        }
    }
}
